// transformerDecoder_72885595013427
// MI455X (gfx1250) — compile-verified, matrix operand order checked by reading, not run on hardware
//
#include <hip/hip_runtime.h>

// ---------------------------------------------------------------------------
// MI455X (gfx1250) fused transformer decoder.
// All GEMMs run on v_wmma_f32_16x16x32_bf16 (bf16 A/B, f32 accumulate).
// One workgroup (256 threads = 8 wave32) per batch element; all layer
// intermediates live in LDS; sig is read once and the output written once.
// ---------------------------------------------------------------------------

#define E_DIM 256
#define S_DIM 18
#define C_DIM 512
#define H_DIM 8
#define DH    32
#define NLAYER 4

typedef __bf16 bf16_t;
typedef __attribute__((ext_vector_type(8)))  bf16_t v8bf;
typedef __attribute__((ext_vector_type(16))) bf16_t v16bf;
typedef __attribute__((ext_vector_type(8)))  float  v8f;

#define V8F_ZERO {0.f,0.f,0.f,0.f,0.f,0.f,0.f,0.f}

// WMMA 16-bit operand fragment (wave32):
//   lane L (<16): row M=L (A) / col N=L (B), holds K = k0..k0+7 and k0+16..k0+23
//   lane L+16   : same row/col, K shifted by 8 (k0 = kh*8 folded in by caller)
// -> two contiguous 16-byte loads per lane.
__device__ __forceinline__ v16bf ldfrag(const bf16_t* p) {
  v8bf lo = *(const v8bf*)(p);
  v8bf hi = *(const v8bf*)(p + 16);
  return __builtin_shufflevector(lo, hi, 0,1,2,3,4,5,6,7,8,9,10,11,12,13,14,15);
}

__device__ __forceinline__ v8f wmma_bf16(v16bf a, v16bf b, v8f c) {
  // D = A(16x32) * B(32x16) + C, f32 accumulate
  return __builtin_amdgcn_wmma_f32_16x16x32_bf16(false, a, false, b, (short)0, c,
                                                 false, false);
}

__device__ __forceinline__ float wred(float v) {
#pragma unroll
  for (int m = 16; m > 0; m >>= 1) v += __shfl_xor(v, m, 32);
  return v;
}

__device__ __forceinline__ float silu_f(float x) { return x / (1.f + __expf(-x)); }

// ---------------------------------------------------------------------------
// Kernel 1: weight bf16 conversion + stats zero + context scalar stash
// ---------------------------------------------------------------------------
__global__ void tdec_prep(const float* __restrict__ xWq, const float* __restrict__ xWv,
                          const float* __restrict__ xWo, const float* __restrict__ W1,
                          const float* __restrict__ W2,  const float* __restrict__ W3,
                          const float* cWq, const float* cWk, const float* cWv,
                          const float* cbq, const float* cbk, const float* cbv,
                          const float* cbias_k, const float* cbias_v,
                          const float* cWo, const float* cbo,
                          bf16_t* __restrict__ wq, bf16_t* __restrict__ wv,
                          bf16_t* __restrict__ wo, bf16_t* __restrict__ w1,
                          bf16_t* __restrict__ w2, bf16_t* __restrict__ w3p,
                          float* __restrict__ stats) {
  int i = blockIdx.x * blockDim.x + threadIdx.x;
  if (i < 64) stats[i] = 0.f;                 // sum[0..31], sumsq[32..63]
  if (i == 64) {
    stats[112] = cWq[0];  stats[113] = cWk[0]; stats[114] = cWv[0];
    stats[115] = cbq[0];  stats[116] = cbk[0]; stats[117] = cbv[0];
    stats[118] = cbias_k[0]; stats[119] = cbias_v[0];
    stats[120] = cWo[0];  stats[121] = cbo[0];
  }
  if (i < E_DIM * E_DIM) {
    wq[i] = (bf16_t)xWq[i];
    wv[i] = (bf16_t)xWv[i];
    wo[i] = (bf16_t)xWo[i];
  }
  if (i < 32 * E_DIM) w1[i] = (bf16_t)W1[i];
  if (i < 16 * 32)    w2[i] = (bf16_t)W2[i];
  if (i < E_DIM * 32) {                       // W3 padded K16 -> K32 with zeros
    int n = i >> 5, k = i & 31;
    w3p[i] = (bf16_t)((k < 16) ? W3[n * 16 + k] : 0.f);
  }
}

// ---------------------------------------------------------------------------
// Kernel 2: ec = ctx @ Ws^T + bs  (B x 512 x 18)  + batch stats accumulation
// ---------------------------------------------------------------------------
__global__ void tdec_ec(const float* __restrict__ ctx, const float* __restrict__ Ws,
                        const float* __restrict__ bs, float* __restrict__ ecRaw,
                        float* __restrict__ stats, int B) {
  int r = blockIdx.x * blockDim.x + threadIdx.x;
  if (r >= B) return;
  float acc[S_DIM];
#pragma unroll
  for (int j = 0; j < S_DIM; ++j) acc[j] = bs[j];
  for (int k = 0; k < C_DIM; k += 4) {
    float c0 = ctx[(size_t)r * C_DIM + k + 0];
    float c1 = ctx[(size_t)r * C_DIM + k + 1];
    float c2 = ctx[(size_t)r * C_DIM + k + 2];
    float c3 = ctx[(size_t)r * C_DIM + k + 3];
#pragma unroll
    for (int j = 0; j < S_DIM; ++j)
      acc[j] += c0 * Ws[j * C_DIM + k] + c1 * Ws[j * C_DIM + k + 1] +
                c2 * Ws[j * C_DIM + k + 2] + c3 * Ws[j * C_DIM + k + 3];
  }
#pragma unroll
  for (int j = 0; j < S_DIM; ++j) {
    ecRaw[(size_t)r * S_DIM + j] = acc[j];
    atomicAdd(&stats[j], acc[j]);
    atomicAdd(&stats[32 + j], acc[j] * acc[j]);
  }
}

// ---------------------------------------------------------------------------
// Kernel 3: finalize batch-norm scale/shift
// ---------------------------------------------------------------------------
__global__ void tdec_bnfin(const float* __restrict__ bn_g, const float* __restrict__ bn_b,
                           float* __restrict__ stats, int B) {
  int j = threadIdx.x;
  if (j < S_DIM) {
    float invB = 1.f / (float)B;
    float mean = stats[j] * invB;
    float var  = stats[32 + j] * invB - mean * mean;
    float sc   = bn_g[j] * rsqrtf(var + 1e-5f);
    stats[64 + j] = sc;                  // scale
    stats[96 + j] = bn_b[j] - mean * sc; // shift
  }
}

// ---------------------------------------------------------------------------
// Kernel 4: fused 4-layer decoder, one block per batch element
// ---------------------------------------------------------------------------
#define SIG_ST 260   // f32 stride for sig rows
#define NS_ST  264   // bf16 stride (mult of 8 halves -> 16B aligned rows)
#define VT_ST  40
#define AP_ST  40
#define H1_ST  40
#define H2_ST  32
#define SP     32    // padded row count (2 M-tiles of 16)

__global__ __launch_bounds__(256) void tdec_main(
    const float* __restrict__ sigIn, const float* __restrict__ ecRaw,
    const float* __restrict__ stats,
    const bf16_t* __restrict__ Wq, const bf16_t* __restrict__ Wv,
    const bf16_t* __restrict__ Wo, const bf16_t* __restrict__ Wf1,
    const bf16_t* __restrict__ Wf2, const bf16_t* __restrict__ Wf3,
    const float* __restrict__ xWk, const float* __restrict__ xbq,
    const float* __restrict__ xbk, const float* __restrict__ xbv,
    const float* __restrict__ xbias_k, const float* __restrict__ xbias_v,
    const float* __restrict__ xbo, const float* __restrict__ b1,
    const float* __restrict__ b2, const float* __restrict__ b3,
    const float* __restrict__ g_cas, const float* __restrict__ bta_cas,
    const float* __restrict__ g_ff,  const float* __restrict__ bta_ff,
    const float* __restrict__ g_fin, const float* __restrict__ bta_fin,
    float* __restrict__ out) {
  __shared__ float sigF[S_DIM * SIG_ST];
  __shared__ float ecF[32];
  __shared__ float scoresF[H_DIM * S_DIM * 20];
  __shared__ __align__(16) bf16_t nsB[SP * NS_ST];   // ns / ao / nf staging
  __shared__ __align__(16) bf16_t qB [SP * NS_ST];
  __shared__ __align__(16) bf16_t kB [SP * NS_ST];
  __shared__ __align__(16) bf16_t vT [E_DIM * VT_ST]; // v transposed [e][kpos]
  __shared__ __align__(16) bf16_t aP [H_DIM * SP * AP_ST];
  __shared__ __align__(16) bf16_t h1 [SP * H1_ST];
  __shared__ __align__(16) bf16_t h2 [SP * H2_ST];

  const int tid  = threadIdx.x;
  const int lane = tid & 31;
  const int wave = tid >> 5;
  const int lm   = lane & 15;  // fragment row/col within tile
  const int kh   = lane >> 4;  // K half select (folds into k0)
  const int b    = blockIdx.x;

  // Zero all bf16 staging once: pad rows/cols stay zero (finite) forever.
  for (int i = tid; i < SP * NS_ST; i += 256) { nsB[i] = (bf16_t)0.f; qB[i] = (bf16_t)0.f; kB[i] = (bf16_t)0.f; }
  for (int i = tid; i < E_DIM * VT_ST; i += 256) vT[i] = (bf16_t)0.f;
  for (int i = tid; i < H_DIM * SP * AP_ST; i += 256) aP[i] = (bf16_t)0.f;
  for (int i = tid; i < SP * H1_ST; i += 256) h1[i] = (bf16_t)0.f;
  for (int i = tid; i < SP * H2_ST; i += 256) h2[i] = (bf16_t)0.f;

  for (int i = tid; i < S_DIM * E_DIM; i += 256) {
    int s = i >> 8, e = i & 255;
    sigF[s * SIG_ST + e] = sigIn[((size_t)b * S_DIM + s) * E_DIM + e];
  }
  if (tid < S_DIM)
    ecF[tid] = ecRaw[(size_t)b * S_DIM + tid] * stats[64 + tid] + stats[96 + tid];

  const float c_wq = stats[112], c_wk = stats[113], c_wv = stats[114];
  const float c_bq = stats[115], c_bk = stats[116], c_bv = stats[117];
  const float c_bkk = stats[118], c_bvv = stats[119];
  const float c_wo = stats[120], c_bo = stats[121];
  const float attScale = 0.17677669529663687f; // 1/sqrt(32)
  __syncthreads();

  for (int layer = 0; layer < NLAYER; ++layer) {
    // ---- (a) context self-attention (feature dim = 1, pure scalar) ----
    float nec = 0.f;
    if (tid < S_DIM) {
      float q = ecF[tid] * c_wq + c_bq;
      float lg[19]; float mx = -1e30f;
#pragma unroll
      for (int k = 0; k < 19; ++k) {
        float kv = (k < S_DIM) ? (ecF[k] * c_wk + c_bk) : c_bkk;
        lg[k] = q * kv; mx = fmaxf(mx, lg[k]);
      }
      float sum = 0.f;
#pragma unroll
      for (int k = 0; k < 19; ++k) { lg[k] = __expf(lg[k] - mx); sum += lg[k]; }
      float inv = 1.f / sum, sa = 0.f;
#pragma unroll
      for (int k = 0; k < 19; ++k) {
        float vv = (k < S_DIM) ? (ecF[k] * c_wv + c_bv) : c_bvv;
        sa += lg[k] * vv;
      }
      nec = ecF[tid] + sa * inv * c_wo + c_bo;
    }
    __syncthreads();
    if (tid < S_DIM) ecF[tid] = nec;
    __syncthreads();

    // ---- (b1) ns = LN_cas(sig) -> nsB(bf16); kB[19xE] from ec; vT bias row ----
    for (int rr = wave; rr < S_DIM; rr += 8) {
      float x[8], sm = 0.f, s2 = 0.f;
#pragma unroll
      for (int i = 0; i < 8; ++i) { x[i] = sigF[rr * SIG_ST + lane + i * 32]; sm += x[i]; s2 += x[i] * x[i]; }
      sm = wred(sm); s2 = wred(s2);
      float mean = sm * (1.f / 256.f);
      float inv = rsqrtf(s2 * (1.f / 256.f) - mean * mean + 1e-10f);
#pragma unroll
      for (int i = 0; i < 8; ++i) {
        int e = lane + i * 32;
        nsB[rr * NS_ST + e] = (bf16_t)((x[i] - mean) * inv * g_cas[e] + bta_cas[e]);
      }
    }
    {
      int e = tid; // 256 threads == E
      float wk = xWk[e], bk = xbk[e];
#pragma unroll
      for (int s = 0; s < S_DIM; ++s) kB[s * NS_ST + e] = (bf16_t)(ecF[s] * wk + bk);
      kB[S_DIM * NS_ST + e] = (bf16_t)xbias_k[e];
      vT[e * VT_ST + S_DIM] = (bf16_t)xbias_v[e];
    }
    __syncthreads();

    // ---- (b2) q & v projections: 64 tiles of 16x16, K=256 ----
    for (int t = wave; t < 64; t += 8) {
      const int which = t >> 5;           // 0=q, 1=v
      const int mt = (t >> 4) & 1, nt = t & 15;
      const bf16_t* W = which ? Wv : Wq;
      v8f acc = V8F_ZERO;
      for (int ks = 0; ks < 8; ++ks) {
        v16bf a  = ldfrag(&nsB[(mt * 16 + lm) * NS_ST + ks * 32 + kh * 8]);
        v16bf bb = ldfrag(&W[(size_t)(nt * 16 + lm) * E_DIM + ks * 32 + kh * 8]);
        acc = wmma_bf16(a, bb, acc);
      }
      const int n = nt * 16 + lm;
      if (which == 0) {
        float bq = xbq[n];
#pragma unroll
        for (int r = 0; r < 8; ++r) {
          int m = mt * 16 + kh * 8 + r;
          qB[m * NS_ST + n] = (bf16_t)((acc[r] + bq) * attScale); // fold 1/sqrt(dh)
        }
      } else {
        float bv = xbv[n];
#pragma unroll
        for (int r = 0; r < 8; ++r) {
          int m = mt * 16 + kh * 8 + r;
          if (m < S_DIM) vT[n * VT_ST + m] = (bf16_t)(acc[r] + bv);
        }
      }
    }
    __syncthreads();

    // ---- (b3) per-head scores = q . k  (K = dh = 32, single WMMA) ----
    for (int t = wave; t < 32; t += 8) {
      const int h = t >> 2, mt = (t >> 1) & 1, nt = t & 1;
      v8f acc = V8F_ZERO;
      v16bf a  = ldfrag(&qB[(mt * 16 + lm) * NS_ST + h * DH + kh * 8]);
      v16bf bb = ldfrag(&kB[(nt * 16 + lm) * NS_ST + h * DH + kh * 8]);
      acc = wmma_bf16(a, bb, acc);
      const int n = nt * 16 + lm;
      if (n < 19) {
#pragma unroll
        for (int r = 0; r < 8; ++r) {
          int m = mt * 16 + kh * 8 + r;
          if (m < S_DIM) scoresF[(h * S_DIM + m) * 20 + n] = acc[r];
        }
      }
    }
    __syncthreads();

    // ---- (b4) softmax over 19 keys (144 rows, one thread each) ----
    if (tid < H_DIM * S_DIM) {
      const int h = tid / S_DIM, s = tid - h * S_DIM;
      const int base = (h * S_DIM + s) * 20;
      float p[19]; float mx = -1e30f;
#pragma unroll
      for (int k = 0; k < 19; ++k) { p[k] = scoresF[base + k]; mx = fmaxf(mx, p[k]); }
      float sum = 0.f;
#pragma unroll
      for (int k = 0; k < 19; ++k) { p[k] = __expf(p[k] - mx); sum += p[k]; }
      float inv = 1.f / sum;
      const int ab = (h * SP + s) * AP_ST;
#pragma unroll
      for (int k = 0; k < 19; ++k) aP[ab + k] = (bf16_t)(p[k] * inv); // K 19..31 stay 0
    }
    __syncthreads();

    // ---- (b5) per-head ctx = probs @ v -> aoB(=nsB reuse) ----
    for (int t = wave; t < 32; t += 8) {
      const int h = t >> 2, mt = (t >> 1) & 1, nt = t & 1;
      v8f acc = V8F_ZERO;
      v16bf a  = ldfrag(&aP[(h * SP + mt * 16 + lm) * AP_ST + kh * 8]);
      v16bf bb = ldfrag(&vT[(h * DH + nt * 16 + lm) * VT_ST + kh * 8]);
      acc = wmma_bf16(a, bb, acc);
      const int n = h * DH + nt * 16 + lm;
#pragma unroll
      for (int r = 0; r < 8; ++r) {
        int m = mt * 16 + kh * 8 + r;
        if (m < S_DIM) nsB[m * NS_ST + n] = (bf16_t)acc[r];
      }
    }
    __syncthreads();

    // ---- (b6) sig += ao @ xWo^T + xbo ----
    for (int t = wave; t < 32; t += 8) {
      const int mt = t >> 4, nt = t & 15;
      v8f acc = V8F_ZERO;
      for (int ks = 0; ks < 8; ++ks) {
        v16bf a  = ldfrag(&nsB[(mt * 16 + lm) * NS_ST + ks * 32 + kh * 8]);
        v16bf bb = ldfrag(&Wo[(size_t)(nt * 16 + lm) * E_DIM + ks * 32 + kh * 8]);
        acc = wmma_bf16(a, bb, acc);
      }
      const int n = nt * 16 + lm;
      const float bo = xbo[n];
#pragma unroll
      for (int r = 0; r < 8; ++r) {
        int m = mt * 16 + kh * 8 + r;
        if (m < S_DIM) sigF[m * SIG_ST + n] += acc[r] + bo;
      }
    }
    __syncthreads();

    // ---- (c1) nf = LN_ff(sig) -> nsB ----
    for (int rr = wave; rr < S_DIM; rr += 8) {
      float x[8], sm = 0.f, s2 = 0.f;
#pragma unroll
      for (int i = 0; i < 8; ++i) { x[i] = sigF[rr * SIG_ST + lane + i * 32]; sm += x[i]; s2 += x[i] * x[i]; }
      sm = wred(sm); s2 = wred(s2);
      float mean = sm * (1.f / 256.f);
      float inv = rsqrtf(s2 * (1.f / 256.f) - mean * mean + 1e-10f);
#pragma unroll
      for (int i = 0; i < 8; ++i) {
        int e = lane + i * 32;
        nsB[rr * NS_ST + e] = (bf16_t)((x[i] - mean) * inv * g_ff[e] + bta_ff[e]);
      }
    }
    __syncthreads();

    // ---- (c2) h1 = silu(nf @ W1^T + b1): N=32, K=256 ----
    for (int t = wave; t < 4; t += 8) {
      const int mt = t >> 1, nt = t & 1;
      v8f acc = V8F_ZERO;
      for (int ks = 0; ks < 8; ++ks) {
        v16bf a  = ldfrag(&nsB[(mt * 16 + lm) * NS_ST + ks * 32 + kh * 8]);
        v16bf bb = ldfrag(&Wf1[(nt * 16 + lm) * E_DIM + ks * 32 + kh * 8]);
        acc = wmma_bf16(a, bb, acc);
      }
      const int n = nt * 16 + lm;
      const float bb1 = b1[n];
#pragma unroll
      for (int r = 0; r < 8; ++r) {
        int m = mt * 16 + kh * 8 + r;
        if (m < S_DIM) h1[m * H1_ST + n] = (bf16_t)silu_f(acc[r] + bb1);
      }
    }
    __syncthreads();

    // ---- (c3) h2 = silu(h1 @ W2^T + b2): N=16, K=32 ----
    for (int t = wave; t < 2; t += 8) {
      const int mt = t;
      v8f acc = V8F_ZERO;
      v16bf a  = ldfrag(&h1[(mt * 16 + lm) * H1_ST + kh * 8]);
      v16bf bb = ldfrag(&Wf2[lm * 32 + kh * 8]);
      acc = wmma_bf16(a, bb, acc);
      const int n = lm;
      const float bb2 = b2[n];
#pragma unroll
      for (int r = 0; r < 8; ++r) {
        int m = mt * 16 + kh * 8 + r;
        if (m < S_DIM) h2[m * H2_ST + n] = (bf16_t)silu_f(acc[r] + bb2); // cols 16..31 stay 0
      }
    }
    __syncthreads();

    // ---- (c4) sig += silu(h2 @ W3^T + b3): N=256, K=16 zero-padded to 32 ----
    for (int t = wave; t < 32; t += 8) {
      const int mt = t >> 4, nt = t & 15;
      v8f acc = V8F_ZERO;
      v16bf a  = ldfrag(&h2[(mt * 16 + lm) * H2_ST + kh * 8]);
      v16bf bb = ldfrag(&Wf3[(nt * 16 + lm) * 32 + kh * 8]);
      acc = wmma_bf16(a, bb, acc);
      const int n = nt * 16 + lm;
      const float bb3 = b3[n];
#pragma unroll
      for (int r = 0; r < 8; ++r) {
        int m = mt * 16 + kh * 8 + r;
        if (m < S_DIM) sigF[m * SIG_ST + n] += silu_f(acc[r] + bb3);
      }
    }
    __syncthreads();
  } // layers

  // ---- final LayerNorm -> out ----
  for (int rr = wave; rr < S_DIM; rr += 8) {
    float x[8], sm = 0.f, s2 = 0.f;
#pragma unroll
    for (int i = 0; i < 8; ++i) { x[i] = sigF[rr * SIG_ST + lane + i * 32]; sm += x[i]; s2 += x[i] * x[i]; }
    sm = wred(sm); s2 = wred(s2);
    float mean = sm * (1.f / 256.f);
    float inv = rsqrtf(s2 * (1.f / 256.f) - mean * mean + 1e-10f);
#pragma unroll
    for (int i = 0; i < 8; ++i) {
      int e = lane + i * 32;
      out[((size_t)b * S_DIM + rr) * E_DIM + e] = (x[i] - mean) * inv * g_fin[e] + bta_fin[e];
    }
  }
}

// ---------------------------------------------------------------------------
// Host launcher
// ---------------------------------------------------------------------------
extern "C" void kernel_launch(void* const* d_in, const int* in_sizes, int n_in,
                              void* d_out, int out_size, void* d_ws, size_t ws_size,
                              hipStream_t stream) {
  const float* sig     = (const float*)d_in[0];
  const float* ctx     = (const float*)d_in[1];
  const float* Ws      = (const float*)d_in[2];
  const float* bs      = (const float*)d_in[3];
  const float* bn_g    = (const float*)d_in[4];
  const float* bn_b    = (const float*)d_in[5];
  const float* cWq     = (const float*)d_in[6];
  const float* cWk     = (const float*)d_in[7];
  const float* cWv     = (const float*)d_in[8];
  const float* cbq     = (const float*)d_in[9];
  const float* cbk     = (const float*)d_in[10];
  const float* cbv     = (const float*)d_in[11];
  const float* cbias_k = (const float*)d_in[12];
  const float* cbias_v = (const float*)d_in[13];
  const float* cWo     = (const float*)d_in[14];
  const float* cbo     = (const float*)d_in[15];
  const float* xWq     = (const float*)d_in[16];
  const float* xWk     = (const float*)d_in[17];
  const float* xWv     = (const float*)d_in[18];
  const float* xbq     = (const float*)d_in[19];
  const float* xbk     = (const float*)d_in[20];
  const float* xbv     = (const float*)d_in[21];
  const float* xbias_k = (const float*)d_in[22];
  const float* xbias_v = (const float*)d_in[23];
  const float* xWo     = (const float*)d_in[24];
  const float* xbo     = (const float*)d_in[25];
  const float* W1      = (const float*)d_in[26];
  const float* b1      = (const float*)d_in[27];
  const float* W2      = (const float*)d_in[28];
  const float* b2      = (const float*)d_in[29];
  const float* W3      = (const float*)d_in[30];
  const float* b3      = (const float*)d_in[31];
  const float* ln_ff_g  = (const float*)d_in[32];
  const float* ln_ff_b  = (const float*)d_in[33];
  const float* ln_cas_g = (const float*)d_in[34];
  const float* ln_cas_b = (const float*)d_in[35];
  const float* ln_fin_g = (const float*)d_in[36];
  const float* ln_fin_b = (const float*)d_in[37];

  const int B = in_sizes[0] / (S_DIM * E_DIM);

  // Workspace layout (all 16B aligned):
  //   [0,512)            stats (sums/scale/shift + context scalars)
  //   [512, +131072)x3   xWq/xWv/xWo bf16
  //   then W1 bf16 (16384), W2 bf16 (1024), W3 padded bf16 (16384)
  //   then ecRaw f32 (B*18)
  char* base = (char*)d_ws;
  float*  ws_stats = (float*)(base);
  bf16_t* ws_wq = (bf16_t*)(base + 512);
  bf16_t* ws_wv = ws_wq + E_DIM * E_DIM;
  bf16_t* ws_wo = ws_wv + E_DIM * E_DIM;
  bf16_t* ws_w1 = ws_wo + E_DIM * E_DIM;
  bf16_t* ws_w2 = ws_w1 + 32 * E_DIM;
  bf16_t* ws_w3 = ws_w2 + 16 * 32;
  float*  ws_ecRaw = (float*)(ws_w3 + E_DIM * 32);

  tdec_prep<<<(E_DIM * E_DIM + 255) / 256, 256, 0, stream>>>(
      xWq, xWv, xWo, W1, W2, W3, cWq, cWk, cWv, cbq, cbk, cbv, cbias_k, cbias_v,
      cWo, cbo, ws_wq, ws_wv, ws_wo, ws_w1, ws_w2, ws_w3, ws_stats);

  tdec_ec<<<(B + 255) / 256, 256, 0, stream>>>(ctx, Ws, bs, ws_ecRaw, ws_stats, B);

  tdec_bnfin<<<1, 32, 0, stream>>>(bn_g, bn_b, ws_stats, B);

  tdec_main<<<B, 256, 0, stream>>>(
      sig, ws_ecRaw, ws_stats, ws_wq, ws_wv, ws_wo, ws_w1, ws_w2, ws_w3,
      xWk, xbq, xbk, xbv, xbias_k, xbias_v, xbo, b1, b2, b3,
      ln_cas_g, ln_cas_b, ln_ff_g, ln_ff_b, ln_fin_g, ln_fin_b, (float*)d_out);
}
